// Unet_CG_40699110097136
// MI455X (gfx1250) — compile-verified, matrix operand order checked by reading, not run on hardware
//
#include <hip/hip_runtime.h>

// ---------------------------------------------------------------------------
// MoDL unrolled recon for MI455X (gfx1250, wave32).
//  - U-Net convs: LDS-staged implicit GEMM on v_wmma_f32_16x16x32_f16
//  - shifted FFT2/IFFT2: DFT-as-GEMM on v_wmma_f32_16x16x4_f32 (full f32)
//  - CG scalars: LDS tree reduction + global f32 atomics (graph-safe)
// ---------------------------------------------------------------------------

typedef _Float16 v16h __attribute__((ext_vector_type(16)));
typedef float    v8f  __attribute__((ext_vector_type(8)));
typedef float    v2f  __attribute__((ext_vector_type(2)));

#define HWF 147456   // 384*384
#define NPX 384

// ---------------- LDS block reduction -> global atomic ---------------------
__device__ __forceinline__ void block_reduce_atomic(float v, float* dst) {
  __shared__ float red[256];
  const int t = threadIdx.x;
  red[t] = v;
  __syncthreads();
#pragma unroll
  for (int s = 128; s > 0; s >>= 1) {
    if (t < s) red[t] += red[t + s];
    __syncthreads();
  }
  if (t == 0) atomicAdd(dst, red[0]);
}

// ---------------- init: shifted DFT matrices Fs, Gs ------------------------
// S.F.S = D.F.D for even N with N/2 even -> Fs[j,k] = (-1)^(j+k) e^{-2pi i jk/N}
__global__ void build_dft(float* __restrict__ FsR, float* __restrict__ FsI,
                          float* __restrict__ GsR, float* __restrict__ GsI) {
  const int idx = blockIdx.x * 256 + threadIdx.x;
  if (idx >= HWF) return;
  const int j = idx / NPX, kc = idx % NPX;
  const int t = (j * kc) % NPX;                      // exact angle reduction
  const float ang = -6.283185307179586f * (float)t / (float)NPX;
  const float s = ((j + kc) & 1) ? -1.f : 1.f;
  const float fr = s * cosf(ang);
  const float fi = s * sinf(ang);
  FsR[idx] = fr;  FsI[idx] = fi;
  GsR[idx] = fr * (1.f / (float)NPX);
  GsI[idx] = -fi * (1.f / (float)NPX);
}

// ---------------- deinterleave [..,2] -> planar re/im ----------------------
__global__ void unpack_ri(const float* __restrict__ src, float* __restrict__ R,
                          float* __restrict__ I, int n) {
  const int i = blockIdx.x * 256 + threadIdx.x;
  if (i < n) { R[i] = src[2 * i]; I[i] = src[2 * i + 1]; }
}

// ---------------- planar -> interleaved complex64 output -------------------
__global__ void store_out(float* __restrict__ dst, const float* __restrict__ R,
                          const float* __restrict__ I, int n) {
  const int i = blockIdx.x * 256 + threadIdx.x;
  if (i < n) { dst[2 * i] = R[i]; dst[2 * i + 1] = I[i]; }
}

// ---------------- 3x3 conv as LDS-staged implicit GEMM (f16 WMMA) ----------
// Activations planar [B][Cin][H][W]; weights [Cout][Cin][3][3].
// K order: k = pos*CIN + cin (pos = ky*3+kx), CIN power of two.
// Block = 128 threads = 4 waves. Block computes a 64-pixel row segment for one
// 16-cout tile; wave w owns pixels [w*16, w*16+16). Input patch and f16
// weights staged in LDS; inner loop is ds_load -> v_wmma only.
template <int CIN0, int CIN, int L2CIN, int HIN, int WIN, int UP,
          int STRIDE, int PAD, int HOUT, int WOUT, int COUT>
__global__ __launch_bounds__(128)
void conv3x3_t(const float* __restrict__ in0, const float* __restrict__ in1,
               const float* __restrict__ w, const float* __restrict__ bias,
               float* __restrict__ out) {
  constexpr int PW   = 64 * STRIDE + 2;   // staged patch width
  constexpr int KTOT = 9 * CIN;
  constexpr int NCH  = (KTOT + 31) / 32;
  constexpr int WS   = UP ? (WIN / 2) : WIN;
  constexpr long HWS = (long)(UP ? (HIN / 2) : HIN) * WS;
  constexpr int NCT  = COUT / 16;

  __shared__ _Float16 patch[CIN * 3 * PW];   // [cin][dy][xo]
  __shared__ _Float16 wlds[16 * KTOT];       // [n][k]

  const int tid = threadIdx.x;
  const int seg = blockIdx.x;
  const int y   = blockIdx.y;
  const int ct  = blockIdx.z % NCT;
  const int b   = blockIdx.z / NCT;

  // ---- stage zero-padded input patch (upsample folded at stage time) ----
  const int iy0 = y * STRIDE - PAD;
  const int ix0 = seg * 64 * STRIDE - PAD;
  for (int idx = tid; idx < CIN * 3 * PW; idx += 128) {
    const int xo  = idx % PW;
    const int r   = idx / PW;
    const int dy  = r % 3;
    const int cin = r / 3;
    const int iy = iy0 + dy;
    const int ix = ix0 + xo;
    float v = 0.f;
    if (iy >= 0 && iy < HIN && ix >= 0 && ix < WIN) {
      const float* src; long pb;
      if (cin < CIN0) { src = in0; pb = ((long)b * CIN0 + cin) * HWS; }
      else { src = in1; pb = ((long)b * (CIN - CIN0) + (cin - CIN0)) * HWS; }
      const long g = UP ? ((long)(iy >> 1) * WS + (ix >> 1)) : ((long)iy * WIN + ix);
      v = src[pb + g];
    }
    patch[idx] = (_Float16)v;
  }
  // ---- stage f16 weights for this cout tile, [n][k] ----
  for (int idx = tid; idx < 16 * KTOT; idx += 128) {
    const int n = idx / KTOT;
    const int k = idx - n * KTOT;
    const int cin = k & (CIN - 1);
    const int pos = k >> L2CIN;
    wlds[idx] = (_Float16)w[((long)(ct * 16 + n) * CIN + cin) * 9 + pos];
  }
  __syncthreads();

  const int lane  = tid & 31;
  const int wv    = tid >> 5;
  const int ln    = lane & 15;
  const int khalf = lane >> 4;
  const int abase = (wv * 16 + ln) * STRIDE;  // A row = this lane's pixel
  const int nbase = ln * KTOT;                // B col = this lane's cout

  v8f acc = {0.f, 0.f, 0.f, 0.f, 0.f, 0.f, 0.f, 0.f};
  for (int kc = 0; kc < NCH; ++kc) {
    v16h af, bf;
#pragma unroll
    for (int e = 0; e < 16; ++e) {
      // 16-bit fragment K pattern (ISA 7.12.2): lanes 0-15 K={0-7,16-23},
      // lanes 16-31 K={8-15,24-31}
      const int pat = ((e >> 3) << 4) + khalf * 8 + (e & 7);
      const int k = kc * 32 + pat;
      _Float16 av = (_Float16)0.f, bv = (_Float16)0.f;
      if ((KTOT % 32 == 0) || (k < KTOT)) {   // folds away except conv1 tail
        const int cin = k & (CIN - 1);
        const int pos = k >> L2CIN;
        const int dy = pos / 3;
        const int dx = pos - dy * 3;
        av = patch[(cin * 3 + dy) * PW + abase + dx];
        bv = wlds[nbase + k];
      }
      af[e] = av;
      bf[e] = bv;
    }
    acc = __builtin_amdgcn_wmma_f32_16x16x32_f16(false, af, false, bf,
                                                 (short)0, acc, false, false);
  }

  // D rows are contiguous output pixels -> 8 consecutive stores per lane
  const float bvs = bias[ct * 16 + ln];
  const long obase = (((long)b * COUT + ct * 16 + ln) * HOUT + y) * WOUT +
                     seg * 64 + wv * 16 + 8 * khalf;
#pragma unroll
  for (int v = 0; v < 8; ++v) {
    out[obase + v] = fmaxf(acc[v] + bvs, 0.f);
  }
}

// ---------------- complex 384x384 GEMM via f32 WMMA ------------------------
// Z = X * Y (complex, planar). sx/sy/sz: per-plane element strides (0=const).
// Optional mask: Z *= mask after accumulate (k-space masking).
__global__ void __launch_bounds__(256)
cgemm384(const float* __restrict__ Xr, const float* __restrict__ Xi, long sx,
         const float* __restrict__ Yr, const float* __restrict__ Yi, long sy,
         float* __restrict__ Zr, float* __restrict__ Zi, long sz,
         int nplanes, const float* __restrict__ mask) {
  const int lane = threadIdx.x & 31;
  const int wid = blockIdx.x * 8 + (threadIdx.x >> 5);
  const int tiles = nplanes * 24 * 24;
  if (wid >= tiles) return;                 // wave-uniform
  int t = wid;
  const int nt = t % 24; t /= 24;
  const int mt = t % 24; t /= 24;
  const int pl = t;
  const int ln = lane & 15;
  const int khalf = lane >> 4;
  const int m = mt * 16 + ln;
  const int n = nt * 16 + ln;

  const float* xr = Xr + (long)pl * sx;
  const float* xi = Xi + (long)pl * sx;
  const float* yr = Yr + (long)pl * sy;
  const float* yi = Yi + (long)pl * sy;
  float* zr = Zr + (long)pl * sz;
  float* zi = Zi + (long)pl * sz;

  v8f crr = {0,0,0,0,0,0,0,0}, cii = {0,0,0,0,0,0,0,0};
  v8f cri = {0,0,0,0,0,0,0,0}, cir = {0,0,0,0,0,0,0,0};

  const long mrow = (long)m * NPX;
  for (int k0 = 0; k0 < NPX; k0 += 4) {
    const int kb = k0 + khalf * 2;          // f32 A 16x4: K split by lane half
    const v2f ar = *(const v2f*)(xr + mrow + kb);      // kb even -> 8B aligned
    const v2f ai = *(const v2f*)(xi + mrow + kb);
    v2f br, bi;
    br[0] = yr[(long)kb * NPX + n];  br[1] = yr[(long)(kb + 1) * NPX + n];
    bi[0] = yi[(long)kb * NPX + n];  bi[1] = yi[(long)(kb + 1) * NPX + n];
    const int kpf = (kb + 16 < NPX) ? (kb + 16) : kb;  // branch-free clamp
    __builtin_prefetch(yr + (long)kpf * NPX + n);
    crr = __builtin_amdgcn_wmma_f32_16x16x4_f32(false, ar, false, br, (short)0, crr, false, false);
    cii = __builtin_amdgcn_wmma_f32_16x16x4_f32(false, ai, false, bi, (short)0, cii, false, false);
    cri = __builtin_amdgcn_wmma_f32_16x16x4_f32(false, ar, false, bi, (short)0, cri, false, false);
    cir = __builtin_amdgcn_wmma_f32_16x16x4_f32(false, ai, false, br, (short)0, cir, false, false);
  }

#pragma unroll
  for (int v = 0; v < 8; ++v) {
    const int mr = mt * 16 + v + 8 * khalf;
    const int nc = nt * 16 + ln;
    float outr = crr[v] - cii[v];
    float outi = cri[v] + cir[v];
    if (mask) {
      const float mk = mask[(long)mr * NPX + nc];
      outr *= mk; outi *= mk;
    }
    zr[(long)mr * NPX + nc] = outr;
    zi[(long)mr * NPX + nc] = outi;
  }
}

// ---------------- coil expand: T[b,c] = csm[b,c] * p[b] --------------------
__global__ void coil_mul(const float* __restrict__ csmR, const float* __restrict__ csmI,
                         const float* __restrict__ pR, const float* __restrict__ pI,
                         float* __restrict__ TR, float* __restrict__ TI, int C, long n) {
  const long idx = (long)blockIdx.x * 256 + threadIdx.x;
  if (idx >= n) return;
  const long plane = idx / HWF;
  const int i = (int)(idx % HWF);
  const int b = (int)(plane / C);
  const float cr = csmR[idx], ci = csmI[idx];
  const float pr = pR[(long)b * HWF + i], pi = pI[(long)b * HWF + i];
  TR[idx] = cr * pr - ci * pi;
  TI[idx] = cr * pi + ci * pr;
}

// ---------------- coil reduce: Ap = sum_c T*conj(csm) + lam*p --------------
__global__ void coil_reduce(const float* __restrict__ TR, const float* __restrict__ TI,
                            const float* __restrict__ csmR, const float* __restrict__ csmI,
                            const float* __restrict__ pR, const float* __restrict__ pI,
                            const float* __restrict__ lam, int kiter, int C,
                            float* __restrict__ ApR, float* __restrict__ ApI) {
  const int i = blockIdx.x * 256 + threadIdx.x;
  const int b = blockIdx.y;
  const long j = (long)b * HWF + i;
  float ar = 0.f, ai = 0.f;
  for (int c = 0; c < C; ++c) {
    const long q = ((long)b * C + c) * HWF + i;
    const float tr = TR[q], ti = TI[q], cr = csmR[q], ci = csmI[q];
    ar += tr * cr + ti * ci;       // Re(T * conj(csm))
    ai += ti * cr - tr * ci;       // Im(T * conj(csm))
  }
  const float l = lam[kiter];
  ApR[j] = ar + l * pR[j];
  ApI[j] = ai + l * pI[j];
}

// ---------------- CG kernels ----------------------------------------------
// scalars S[b*4+0]=rTr, +1=pAp, +2=rTrn
__global__ void zero_scalars(float* __restrict__ S) {
  if (threadIdx.x < 16) S[threadIdx.x] = 0.f;
}

__global__ void cg_init(const float* __restrict__ rhsR, const float* __restrict__ rhsI,
                        float* __restrict__ xR, float* __restrict__ xI,
                        float* __restrict__ rR, float* __restrict__ rI,
                        float* __restrict__ pR, float* __restrict__ pI,
                        float* __restrict__ S) {
  const int b = blockIdx.y;
  const long j = (long)b * HWF + blockIdx.x * 256 + threadIdx.x;
  const float vr = rhsR[j], vi = rhsI[j];
  xR[j] = 0.f; xI[j] = 0.f;
  rR[j] = vr;  rI[j] = vi;
  pR[j] = vr;  pI[j] = vi;
  block_reduce_atomic(vr * vr + vi * vi, &S[b * 4 + 0]);
}

__global__ void dot_pAp(const float* __restrict__ pR, const float* __restrict__ pI,
                        const float* __restrict__ ApR, const float* __restrict__ ApI,
                        float* __restrict__ S) {
  const int b = blockIdx.y;
  const long j = (long)b * HWF + blockIdx.x * 256 + threadIdx.x;
  block_reduce_atomic(pR[j] * ApR[j] + pI[j] * ApI[j], &S[b * 4 + 1]);
}

__global__ void cg_update_xr(float* __restrict__ xR, float* __restrict__ xI,
                             float* __restrict__ rR, float* __restrict__ rI,
                             const float* __restrict__ pR, const float* __restrict__ pI,
                             const float* __restrict__ ApR, const float* __restrict__ ApI,
                             float* __restrict__ S) {
  const int b = blockIdx.y;
  const long j = (long)b * HWF + blockIdx.x * 256 + threadIdx.x;
  const float alpha = S[b * 4 + 0] / S[b * 4 + 1];
  xR[j] += alpha * pR[j];
  xI[j] += alpha * pI[j];
  const float nr = rR[j] - alpha * ApR[j];
  const float ni = rI[j] - alpha * ApI[j];
  rR[j] = nr; rI[j] = ni;
  block_reduce_atomic(nr * nr + ni * ni, &S[b * 4 + 2]);
}

__global__ void cg_update_p(float* __restrict__ pR, float* __restrict__ pI,
                            const float* __restrict__ rR, const float* __restrict__ rI,
                            const float* __restrict__ S) {
  const int b = blockIdx.y;
  const long j = (long)b * HWF + blockIdx.x * 256 + threadIdx.x;
  const float beta = S[b * 4 + 2] / S[b * 4 + 0];
  pR[j] = rR[j] + beta * pR[j];
  pI[j] = rI[j] + beta * pI[j];
}

__global__ void scalar_step(float* __restrict__ S, int B) {
  const int b = threadIdx.x;
  if (b < B) { S[b * 4 + 0] = S[b * 4 + 2]; S[b * 4 + 1] = 0.f; S[b * 4 + 2] = 0.f; }
}

// -------- 1x1 output conv + pred store + rhs = atb_dc + lam*pred -----------
__global__ void convout_rhs(const float* __restrict__ h4,
                            const float* __restrict__ wo, const float* __restrict__ bo,
                            const float* __restrict__ lam, int kiter,
                            const float* __restrict__ adcR, const float* __restrict__ adcI,
                            float* __restrict__ predOut,   // interleaved d_out slot
                            float* __restrict__ rhsR, float* __restrict__ rhsI) {
  const int i = blockIdx.x * 256 + threadIdx.x;
  const int b = blockIdx.y;
  const long j = (long)b * HWF + i;
  float a0 = bo[0], a1 = bo[1];
  for (int c = 0; c < 32; ++c) {
    const float h = h4[((long)b * 32 + c) * HWF + i];
    a0 += h * wo[c];
    a1 += h * wo[32 + c];
  }
  predOut[2 * j] = a0;
  predOut[2 * j + 1] = a1;
  const float l = lam[kiter];
  rhsR[j] = adcR[j] + l * a0;
  rhsI[j] = adcI[j] + l * a1;
}

// ===========================================================================
extern "C" void kernel_launch(void* const* d_in, const int* in_sizes, int n_in,
                              void* d_out, int out_size, void* d_ws, size_t ws_size,
                              hipStream_t stream) {
  (void)in_sizes; (void)n_in; (void)out_size; (void)ws_size;
  const int B = 2, C = 12, HW = HWF, HW2 = 192 * 192, NIT = 3, CGI = 20;

  const float* atb_ri = (const float*)d_in[0];
  const float* adc_ri = (const float*)d_in[1];
  const float* csm_ri = (const float*)d_in[2];
  const float* mask   = (const float*)d_in[3];
  const float* lam    = (const float*)d_in[4];
  const float* w1 = (const float*)d_in[5];  const float* b1 = (const float*)d_in[6];
  const float* w2 = (const float*)d_in[7];  const float* b2 = (const float*)d_in[8];
  const float* w3 = (const float*)d_in[9];  const float* b3 = (const float*)d_in[10];
  const float* wup = (const float*)d_in[11]; const float* bup = (const float*)d_in[12];
  const float* w4 = (const float*)d_in[13]; const float* b4 = (const float*)d_in[14];
  const float* wo = (const float*)d_in[15]; const float* bo = (const float*)d_in[16];

  float* ws = (float*)d_ws;
  size_t off = 0;
  auto alloc = [&](size_t n) { float* p = ws + off; off += n; return p; };

  float* FsR = alloc(HW); float* FsI = alloc(HW);
  float* GsR = alloc(HW); float* GsI = alloc(HW);
  float* csmR = alloc((size_t)B * C * HW); float* csmI = alloc((size_t)B * C * HW);
  float* atbR = alloc((size_t)B * HW); float* atbI = alloc((size_t)B * HW);
  float* adcR = alloc((size_t)B * HW); float* adcI = alloc((size_t)B * HW);
  float* xcurR = alloc((size_t)B * HW); float* xcurI = alloc((size_t)B * HW);
  float* rhsR = alloc((size_t)B * HW); float* rhsI = alloc((size_t)B * HW);
  float* xR = alloc((size_t)B * HW); float* xI = alloc((size_t)B * HW);
  float* rR = alloc((size_t)B * HW); float* rI = alloc((size_t)B * HW);
  float* pR = alloc((size_t)B * HW); float* pI = alloc((size_t)B * HW);
  float* ApR = alloc((size_t)B * HW); float* ApI = alloc((size_t)B * HW);
  float* S = alloc(16);
  float* h1 = alloc((size_t)B * 32 * HW);    // also aliased as coil buffer A
  float* h2 = alloc((size_t)B * 64 * HW2);
  float* h3 = alloc((size_t)B * 64 * HW2);
  float* hu = alloc((size_t)B * 32 * HW);    // also aliased as coil buffer B
  float* h4 = alloc((size_t)B * 32 * HW);
  // coil ping-pong buffers alias U-Net activations (disjoint lifetimes)
  float* AbR = h1; float* AbI = h1 + (size_t)B * C * HW;
  float* BbR = hu; float* BbI = hu + (size_t)B * C * HW;

  float* outp = (float*)d_out;               // 7 slots of B*HW complex64

  // ---- init -----
  build_dft<<<576, 256, 0, stream>>>(FsR, FsI, GsR, GsI);
  unpack_ri<<<(B * HW + 255) / 256, 256, 0, stream>>>(atb_ri, atbR, atbI, B * HW);
  unpack_ri<<<(B * HW + 255) / 256, 256, 0, stream>>>(adc_ri, adcR, adcI, B * HW);
  unpack_ri<<<(B * C * HW + 255) / 256, 256, 0, stream>>>(csm_ri, csmR, csmI, B * C * HW);
  hipMemcpyAsync(xcurR, atbR, sizeof(float) * B * HW, hipMemcpyDeviceToDevice, stream);
  hipMemcpyAsync(xcurI, atbI, sizeof(float) * B * HW, hipMemcpyDeviceToDevice, stream);
  store_out<<<(B * HW + 255) / 256, 256, 0, stream>>>(outp, atbR, atbI, B * HW);

  auto cgemm = [&](const float* XR, const float* XI, long sx,
                   const float* YR, const float* YI, long sy,
                   float* ZR, float* ZI, const float* mk) {
    const int tiles = B * C * 24 * 24;
    cgemm384<<<(tiles + 7) / 8, 256, 0, stream>>>(XR, XI, sx, YR, YI, sy,
                                                  ZR, ZI, (long)HW, B * C, mk);
  };
  const dim3 gBHW(576, B);

  for (int k = 0; k < NIT; ++k) {
    // ---------------- U-Net ----------------
    // conv1: Cin=2 (re plane + im plane as split inputs), 384x384, s1 p1
    conv3x3_t<1, 2, 1, 384, 384, 0, 1, 1, 384, 384, 32>
        <<<dim3(6, 384, B * 2), 128, 0, stream>>>(
            xcurR, xcurI, w1 + (size_t)k * 576, b1 + (size_t)k * 32, h1);
    // conv2: 32->64, stride 2 (SAME: pad_low=0), 384->192
    conv3x3_t<32, 32, 5, 384, 384, 0, 2, 0, 192, 192, 64>
        <<<dim3(3, 192, B * 4), 128, 0, stream>>>(
            h1, nullptr, w2 + (size_t)k * 18432, b2 + (size_t)k * 64, h2);
    // conv3: 64->64, 192x192, s1 p1
    conv3x3_t<64, 64, 6, 192, 192, 0, 1, 1, 192, 192, 64>
        <<<dim3(3, 192, B * 4), 128, 0, stream>>>(
            h2, nullptr, w3 + (size_t)k * 36864, b3 + (size_t)k * 64, h3);
    // convup: fused nearest-2x upsample, 64->32, out 384x384
    conv3x3_t<64, 64, 6, 384, 384, 1, 1, 1, 384, 384, 32>
        <<<dim3(6, 384, B * 2), 128, 0, stream>>>(
            h3, nullptr, wup + (size_t)k * 18432, bup + (size_t)k * 32, hu);
    // conv4: fused concat [u, h1], 64->32, 384x384
    conv3x3_t<32, 64, 6, 384, 384, 0, 1, 1, 384, 384, 32>
        <<<dim3(6, 384, B * 2), 128, 0, stream>>>(
            hu, h1, w4 + (size_t)k * 18432, b4 + (size_t)k * 32, h4);
    convout_rhs<<<gBHW, 256, 0, stream>>>(h4, wo + (size_t)k * 64, bo + (size_t)k * 2,
                                          lam, k, adcR, adcI,
                                          outp + (size_t)(4 + k) * B * HW * 2,
                                          rhsR, rhsI);

    // ---------------- CG (20 fixed iterations) ----------------
    zero_scalars<<<1, 32, 0, stream>>>(S);
    cg_init<<<gBHW, 256, 0, stream>>>(rhsR, rhsI, xR, xI, rR, rI, pR, pI, S);
    for (int it = 0; it < CGI; ++it) {
      coil_mul<<<(B * C * HW + 255) / 256, 256, 0, stream>>>(
          csmR, csmI, pR, pI, AbR, AbI, C, (long)B * C * HW);
      cgemm(FsR, FsI, 0, AbR, AbI, HW, BbR, BbI, nullptr);   // rows:  Fs * T
      cgemm(BbR, BbI, HW, FsR, FsI, 0, AbR, AbI, mask);      // cols: (.)*Fs, *mask
      cgemm(GsR, GsI, 0, AbR, AbI, HW, BbR, BbI, nullptr);   // rows:  Gs * K
      cgemm(BbR, BbI, HW, GsR, GsI, 0, AbR, AbI, nullptr);   // cols: (.)*Gs
      coil_reduce<<<gBHW, 256, 0, stream>>>(AbR, AbI, csmR, csmI, pR, pI,
                                            lam, k, C, ApR, ApI);
      dot_pAp<<<gBHW, 256, 0, stream>>>(pR, pI, ApR, ApI, S);
      cg_update_xr<<<gBHW, 256, 0, stream>>>(xR, xI, rR, rI, pR, pI, ApR, ApI, S);
      cg_update_p<<<gBHW, 256, 0, stream>>>(pR, pI, rR, rI, S);
      scalar_step<<<1, 32, 0, stream>>>(S, B);
    }

    store_out<<<(B * HW + 255) / 256, 256, 0, stream>>>(
        outp + (size_t)(k + 1) * B * HW * 2, xR, xI, B * HW);
    hipMemcpyAsync(xcurR, xR, sizeof(float) * B * HW, hipMemcpyDeviceToDevice, stream);
    hipMemcpyAsync(xcurI, xI, sizeof(float) * B * HW, hipMemcpyDeviceToDevice, stream);
  }
}
